// DGCNNSeg_5609227289195
// MI455X (gfx1250) — matrix operand order repaired and checked by reading, not compiled, not run on hardware
//
#include <hip/hip_runtime.h>
#include <hip/hip_bf16.h>

typedef __attribute__((ext_vector_type(16))) _Float16 v16h;
typedef __attribute__((ext_vector_type(8)))  _Float16 v8h;
typedef __attribute__((ext_vector_type(8)))  float    v8f;

#define LEAK 0.2f
#define KTOP 20

__device__ __forceinline__ float leakyf(float v) { return v > 0.f ? v : LEAK * v; }

// ---------------------------------------------------------------------------
// CDNA5 async memory->LDS copy (16 bytes per lane), tracked by ASYNCcnt.
// Generic pointers to __shared__ carry the LDS byte offset in their low 32
// bits (flat aperture rules), which is exactly the VDST operand.
// ---------------------------------------------------------------------------
__device__ __forceinline__ void async_ld16(void* lds_dst, const void* gsrc)
{
    const unsigned off = (unsigned)(uintptr_t)lds_dst;
    asm volatile("global_load_async_to_lds_b128 %0, %1, off"
                 :: "v"(off), "v"(gsrc)
                 : "memory");
}
// wait until ASYNCcnt <= N (N = size of the in-flight prefetch group)
template <int N>
__device__ __forceinline__ void async_wait_le()
{
    asm volatile("s_wait_asynccnt %0" :: "i"(N) : "memory");
}

__device__ __forceinline__ v16h frag16(const _Float16* lo, const _Float16* hi8)
{
    v8h a = *(const v8h*)lo;
    v8h b = *(const v8h*)hi8;
    return __builtin_shufflevector(a, b, 0, 1, 2, 3, 4, 5, 6, 7,
                                         8, 9, 10, 11, 12, 13, 14, 15);
}

// ---------------------------------------------------------------------------
// One wave computes a 16x64 tile of Y = X * W^T (4 WMMAs per 32-K chunk,
// shared A fragment).  f16 operands, async-LDS staging, double-buffered so
// the next chunk's copies overlap the current chunk's WMMAs.
// MODE 0: leaky(y*s+b) -> OutT;  MODE 1: y+b -> OutT;  MODE 2: raw -> OutT.
// ---------------------------------------------------------------------------
template <int MODE, typename OutT>
__global__ __launch_bounds__(32)
void gemm16x64_wmma(const _Float16* __restrict__ X, int ldx, int xcol0,
                    const _Float16* __restrict__ W, int ldw,
                    OutT* __restrict__ Y, int ldy, int ycol0,
                    const float* __restrict__ scale, const float* __restrict__ bias,
                    int K, int Oact)
{
    __shared__ _Float16 As[2][16 * 32];   // [row][k]
    __shared__ _Float16 Bs[2][64 * 32];   // [n][k]

    const int lane = threadIdx.x;
    const bool hi  = lane >= 16;
    const int l    = lane & 15;
    const int m0   = blockIdx.x * 16;
    const int o0   = blockIdx.y * 64;

    // issue one chunk's staging: 2 (A) + 8 (B) = 10 async instructions
    auto issue = [&](int buf, int kb) {
        const int row = lane >> 1, part = lane & 1;
        const _Float16* ga = X + (size_t)(m0 + row) * ldx + xcol0 + kb + part * 16;
        async_ld16(&As[buf][row * 32 + part * 16], ga);
        async_ld16(&As[buf][row * 32 + part * 16 + 8], ga + 8);
        #pragma unroll
        for (int t = 0; t < 8; ++t) {
            const int id = lane + t * 32;     // 0..255 16B chunks
            const int n  = id >> 2;
            const int p  = id & 3;
            const int o  = o0 + n;
            const int oc = (o < Oact) ? o : 0;
            async_ld16(&Bs[buf][n * 32 + p * 8], W + (size_t)oc * ldw + kb + p * 8);
        }
    };

    v8f acc0 = {}, acc1 = {}, acc2 = {}, acc3 = {};
    const int nchunk = K >> 5;
    issue(0, 0);
    for (int ch = 0; ch < nchunk; ++ch) {
        const int cur = ch & 1;
        if (ch + 1 < nchunk) { issue(1 - cur, (ch + 1) << 5); async_wait_le<10>(); }
        else                 { async_wait_le<0>(); }
        __syncthreads();
        const _Float16* A = As[cur];
        const _Float16* Bt = Bs[cur];
        const int ak = hi ? 16 : 0;
        const v16h a = frag16(&A[l * 32 + ak], &A[l * 32 + ak + 8]);
        const int bk = hi ? 16 : 0;
        const v16h b0 = frag16(&Bt[(l +  0) * 32 + bk], &Bt[(l +  0) * 32 + bk + 8]);
        const v16h b1 = frag16(&Bt[(l + 16) * 32 + bk], &Bt[(l + 16) * 32 + bk + 8]);
        const v16h b2 = frag16(&Bt[(l + 32) * 32 + bk], &Bt[(l + 32) * 32 + bk + 8]);
        const v16h b3 = frag16(&Bt[(l + 48) * 32 + bk], &Bt[(l + 48) * 32 + bk + 8]);
        acc0 = __builtin_amdgcn_wmma_f32_16x16x32_f16(false, a, false, b0, (short)0, acc0, false, false);
        acc1 = __builtin_amdgcn_wmma_f32_16x16x32_f16(false, a, false, b1, (short)0, acc1, false, false);
        acc2 = __builtin_amdgcn_wmma_f32_16x16x32_f16(false, a, false, b2, (short)0, acc2, false, false);
        acc3 = __builtin_amdgcn_wmma_f32_16x16x32_f16(false, a, false, b3, (short)0, acc3, false, false);
        __syncthreads();
    }

    // ---- epilogue ----
    const int mrow = m0 + (hi ? 8 : 0);
    #pragma unroll
    for (int s = 0; s < 4; ++s) {
        const v8f acc = (s == 0) ? acc0 : (s == 1) ? acc1 : (s == 2) ? acc2 : acc3;
        const int o = o0 + s * 16 + l;
        float sc = 1.f, bi = 0.f;
        if (MODE == 0) { if (o < Oact) { sc = scale[o]; bi = bias[o]; } }
        if (MODE == 1) { if (o < Oact) { bi = bias[o]; } }
        #pragma unroll
        for (int r = 0; r < 8; ++r) {
            float v = acc[r];
            if (MODE == 0) v = leakyf(v * sc + bi);
            if (MODE == 1) v = v + bi;
            if (o < Oact) Y[(size_t)(mrow + r) * ldy + ycol0 + o] = (OutT)v;
        }
    }
}

// ---------------------------------------------------------------------------
// Fused KNN on f16 features (K = NCHUNK*32).  One wave owns 16 query points.
// Candidate tiles are staged full-depth via async-LDS, double-buffered across
// column tiles; the inner loop is a pure ds_load_b128 + WMMA stream.
// ---------------------------------------------------------------------------
template <int NCHUNK>
__global__ __launch_bounds__(32)
void knn_wmma(const _Float16* __restrict__ X, int ldx, int xcol0,
              const float* __restrict__ sq, int* __restrict__ nidx, int N)
{
    constexpr int K = NCHUNK * 32;
    constexpr int G = 2 * NCHUNK;          // async instructions per tile group
    __shared__ _Float16 As[16 * 128];      // query rows [row][k], stride 128
    __shared__ _Float16 Bs[2][16 * 128];   // candidate tiles [n][k], stride 128
    __shared__ float dtile[16 * 17];
    __shared__ float sqs[16];
    __shared__ float topD[16 * KTOP];
    __shared__ int   topI[16 * KTOP];

    const int lane = threadIdx.x;
    const bool hi  = lane >= 16;
    const int l    = lane & 15;
    const int b    = blockIdx.y;
    const int i0   = blockIdx.x * 16;
    const _Float16* Xb = X + (size_t)b * N * ldx;

    // preload the 16 query rows (G async instructions)
    {
        const int row = lane >> 1, part = lane & 1;
        #pragma unroll
        for (int c = 0; c < NCHUNK; ++c) {
            const int kb = c * 32;
            const _Float16* g = Xb + (size_t)(i0 + row) * ldx + xcol0 + kb + part * 16;
            async_ld16(&As[row * 128 + kb + part * 16], g);
            async_ld16(&As[row * 128 + kb + part * 16 + 8], g + 8);
        }
    }
    // stage one candidate tile full-depth (G async instructions)
    auto issueB = [&](int buf, int j0) {
        const int k0 = hi ? 16 : 0;
        #pragma unroll
        for (int c = 0; c < NCHUNK; ++c) {
            const int kb = c * 32;
            const _Float16* g = Xb + (size_t)(j0 + l) * ldx + xcol0 + kb + k0;
            async_ld16(&Bs[buf][l * 128 + kb + k0], g);
            async_ld16(&Bs[buf][l * 128 + kb + k0 + 8], g + 8);
        }
    };

    if (lane < 16) {
        sqs[lane] = sq[b * N + i0 + lane];
        for (int t = 0; t < KTOP; ++t) {
            topD[lane * KTOP + t] = 3.4e38f;
            topI[lane * KTOP + t] = 0;
        }
    }
    issueB(0, 0);

    const int ntiles = N / 16;
    for (int ct = 0; ct < ntiles; ++ct) {
        const int cur = ct & 1;
        if (ct + 1 < ntiles) { issueB(1 - cur, (ct + 1) * 16); async_wait_le<G>(); }
        else                 { async_wait_le<0>(); }
        __syncthreads();
        const _Float16* Bt = Bs[cur];
        v8f acc = {};
        const int ak = hi ? 16 : 0;
        const int bk = hi ? 16 : 0;
        #pragma unroll
        for (int c = 0; c < NCHUNK; ++c) {
            const int kb = c * 32;
            const v16h a  = frag16(&As[l * 128 + kb + ak], &As[l * 128 + kb + ak + 8]);
            const v16h bb = frag16(&Bt[l * 128 + kb + bk], &Bt[l * 128 + kb + bk + 8]);
            acc = __builtin_amdgcn_wmma_f32_16x16x32_f16(false, a, false, bb,
                                                         (short)0, acc, false, false);
        }
        // d(i,j) = |xi|^2 + |xj|^2 - 2 xi.xj   (lane holds column l, 8 rows)
        const int j0 = ct * 16;
        const float sqj = sq[b * N + j0 + l];
        #pragma unroll
        for (int r = 0; r < 8; ++r) {
            const int row = r + (hi ? 8 : 0);
            dtile[row * 17 + l] = sqs[row] + sqj - 2.0f * acc[r];
        }
        __syncthreads();
        if (lane < 16) {
            float* td = &topD[lane * KTOP];
            int*   ti = &topI[lane * KTOP];
            #pragma unroll 1
            for (int n = 0; n < 16; ++n) {
                const float dd = dtile[lane * 17 + n];
                if (dd < td[KTOP - 1]) {
                    int p = KTOP - 1;
                    while (p > 0 && td[p - 1] > dd) {
                        td[p] = td[p - 1]; ti[p] = ti[p - 1]; --p;
                    }
                    td[p] = dd; ti[p] = j0 + n;
                }
            }
        }
        __syncthreads();
    }
    if (lane < 16) {
        int* dst = nidx + (size_t)(b * N + i0 + lane) * KTOP;
        for (int t = 0; t < KTOP; ++t) dst[t] = topI[lane * KTOP + t];
    }
}

// ---------------------------------------------------------------------------
// Helpers
// ---------------------------------------------------------------------------
__global__ void f32_to_f16_kernel(const float* __restrict__ s,
                                  _Float16* __restrict__ d, int total)
{
    const int t = blockIdx.x * blockDim.x + threadIdx.x;
    if (t < total) d[t] = (_Float16)s[t];
}

__global__ void pad_xyz_kernel(const float* __restrict__ xyz,
                               _Float16* __restrict__ xp, int total /* M*32 */)
{
    const int t = blockIdx.x * blockDim.x + threadIdx.x;
    if (t >= total) return;
    const int c = t & 31;
    const int i = t >> 5;
    xp[t] = (c < 3) ? (_Float16)xyz[i * 3 + c] : (_Float16)0.f;
}

__global__ void sqnorm_kernel(const _Float16* __restrict__ X, int ldx, int xcol0,
                              float* __restrict__ sq, int total, int K)
{
    const int i = blockIdx.x * blockDim.x + threadIdx.x;
    if (i >= total) return;
    const _Float16* src = X + (size_t)i * ldx + xcol0;
    float s = 0.f;
    for (int c = 0; c < K; ++c) { const float v = (float)src[c]; s += v * v; }
    sq[i] = s;
}

// w: [O, 2C] f32 -> ws: [2O, Cpad] f16 zero-padded, rows [W1 ; W2-W1]
__global__ void make_wsplit(const float* __restrict__ w, _Float16* __restrict__ ws,
                            int O, int C, int Cpad)
{
    const int t = blockIdx.x * blockDim.x + threadIdx.x;
    if (t >= 2 * O * Cpad) return;
    const int r = t / Cpad, c = t % Cpad;
    float v = 0.f;
    if (c < C)
        v = (r < O) ? w[(size_t)r * 2 * C + c]
                    : (w[(size_t)(r - O) * 2 * C + C + c]
                       - w[(size_t)(r - O) * 2 * C + c]);
    ws[t] = (_Float16)v;
}

// out[i,o] = max_t leaky((Y1[j_t,o] + T[i,o])*s[o] + b[o])   (f32 in, f16 out)
__global__ void edge_reduce(const float* __restrict__ ycomb,
                            const int* __restrict__ nidx,
                            const float* __restrict__ scale,
                            const float* __restrict__ bias,
                            _Float16* __restrict__ out, int ldo, int ycol0,
                            int Mrows, int O, int N)
{
    const int t = blockIdx.x * blockDim.x + threadIdx.x;
    if (t >= Mrows * O) return;
    const int o = t % O;
    const int i = t / O;
    const int b = i / N;
    const size_t base = (size_t)b * N;
    const float ti = ycomb[(size_t)i * (2 * O) + O + o];
    const int* nb  = nidx + (size_t)i * KTOP;
    const float s = scale[o], bi = bias[o];
    float m = -3.4e38f;
    #pragma unroll 1
    for (int e = 0; e < KTOP; ++e) {
        const int j = nb[e];
        const float y = leakyf((ycomb[(base + j) * (size_t)(2 * O) + o] + ti) * s + bi);
        m = fmaxf(m, y);
    }
    out[(size_t)i * ldo + ycol0 + o] = (_Float16)m;
}

__global__ void colmax_kernel(const _Float16* __restrict__ X, _Float16* __restrict__ g,
                              int N, int C)
{
    const int c = blockIdx.x * blockDim.x + threadIdx.x;
    const int b = blockIdx.y;
    if (c >= C) return;
    const _Float16* src = X + (size_t)b * N * C + c;
    float m = -3.4e38f;
    for (int n = 0; n < N; ++n) m = fmaxf(m, (float)src[(size_t)n * C]);
    g[b * C + c] = (_Float16)m;
}

__global__ void bcast_kernel(const _Float16* __restrict__ g, _Float16* __restrict__ xfin,
                             int N, int ldf, int col0, int C, int total)
{
    const int t = blockIdx.x * blockDim.x + threadIdx.x;
    if (t >= total) return;
    const int c = t % C;
    const int i = t / C;
    const int b = i / N;
    xfin[(size_t)i * ldf + col0 + c] = g[b * C + c];
}

// ---------------------------------------------------------------------------
extern "C" void kernel_launch(void* const* d_in, const int* in_sizes, int n_in,
                              void* d_out, int out_size, void* d_ws, size_t ws_size,
                              hipStream_t stream)
{
    const int B = 4, N = 4096, M = B * N;

    const float* xyz = (const float*)d_in[0];
    const float* w1  = (const float*)d_in[2];
    const float* s1  = (const float*)d_in[3];
    const float* b1  = (const float*)d_in[4];
    const float* w2  = (const float*)d_in[5];
    const float* s2  = (const float*)d_in[6];
    const float* b2  = (const float*)d_in[7];
    const float* w3  = (const float*)d_in[8];
    const float* s3  = (const float*)d_in[9];
    const float* b3  = (const float*)d_in[10];
    const float* w4  = (const float*)d_in[11];
    const float* s4  = (const float*)d_in[12];
    const float* b4  = (const float*)d_in[13];
    const float* wf  = (const float*)d_in[14];
    const float* sf  = (const float*)d_in[15];
    const float* bf  = (const float*)d_in[16];
    const float* we  = (const float*)d_in[17];
    const float* se  = (const float*)d_in[18];
    const float* be  = (const float*)d_in[19];
    const float* wh1 = (const float*)d_in[20];
    const float* sh1 = (const float*)d_in[21];
    const float* bh1 = (const float*)d_in[22];
    const float* wh2 = (const float*)d_in[23];
    const float* sh2 = (const float*)d_in[24];
    const float* bh2 = (const float*)d_in[25];
    const float* wh3 = (const float*)d_in[26];
    const float* bh3 = (const float*)d_in[27];
    float* logits = (float*)d_out;

    // workspace layout
    char* ws = (char*)d_ws;
    size_t off = 0;
    auto alloc = [&](size_t bytes) -> void* {
        void* p = ws + off;
        off = (off + bytes + 255) & ~(size_t)255;
        return p;
    };
    float*     sq      = (float*)    alloc((size_t)M * 4);
    _Float16*  xyzpad  = (_Float16*) alloc((size_t)M * 32 * 2);
    int*       nidx    = (int*)      alloc((size_t)M * KTOP * 4);
    _Float16*  wsplit  = (_Float16*) alloc((size_t)512 * 256 * 2);
    float*     ycomb   = (float*)    alloc((size_t)M * 512 * 4);
    _Float16*  xcat    = (_Float16*) alloc((size_t)M * 512 * 2);
    _Float16*  xfin    = (_Float16*) alloc((size_t)M * 1536 * 2);
    _Float16*  xemb    = (_Float16*) alloc((size_t)M * 1024 * 2);
    _Float16*  glob    = (_Float16*) alloc((size_t)B * 1024 * 2);
    _Float16*  wf16    = (_Float16*) alloc((size_t)512 * 512 * 2);
    _Float16*  we16    = (_Float16*) alloc((size_t)1024 * 512 * 2);
    _Float16*  wh116   = (_Float16*) alloc((size_t)256 * 1536 * 2);
    _Float16*  wh216   = (_Float16*) alloc((size_t)256 * 256 * 2);
    _Float16*  wh316   = (_Float16*) alloc((size_t)50 * 256 * 2);
    _Float16*  h1 = xemb;                 // reuse: xemb dead after colmax/bcast
    _Float16*  h2 = (_Float16*)ycomb;     // reuse: ycomb dead after layer 4

    auto cvt = [&](const float* s, _Float16* d, int n) {
        f32_to_f16_kernel<<<(n + 255) / 256, 256, 0, stream>>>(s, d, n);
    };
    auto gemm_grid = [&](int O) { return dim3(M / 16, (O + 63) / 64); };

    auto edge_layer = [&](const _Float16* Xp, int ldx, int xcol0, int C, int Kpad,
                          const float* w, const float* s, const float* bv,
                          int O, int ycol0) {
        sqnorm_kernel<<<(M + 255) / 256, 256, 0, stream>>>(Xp, ldx, xcol0, sq, M, Kpad);
        const dim3 kg(N / 16, B);
        if (Kpad == 32)
            knn_wmma<1><<<kg, 32, 0, stream>>>(Xp, ldx, xcol0, sq, nidx, N);
        else if (Kpad == 64)
            knn_wmma<2><<<kg, 32, 0, stream>>>(Xp, ldx, xcol0, sq, nidx, N);
        else
            knn_wmma<4><<<kg, 32, 0, stream>>>(Xp, ldx, xcol0, sq, nidx, N);
        const int wtot = 2 * O * Kpad;
        make_wsplit<<<(wtot + 255) / 256, 256, 0, stream>>>(w, wsplit, O, C, Kpad);
        gemm16x64_wmma<2, float><<<gemm_grid(2 * O), 32, 0, stream>>>(
            Xp, ldx, xcol0, wsplit, Kpad, ycomb, 2 * O, 0, nullptr, nullptr, Kpad, 2 * O);
        const int etot = M * O;
        edge_reduce<<<(etot + 255) / 256, 256, 0, stream>>>(
            ycomb, nidx, s, bv, xcat, 512, ycol0, M, O, N);
    };

    // weight f16 mirrors + zero-padded f16 xyz
    pad_xyz_kernel<<<(M * 32 + 255) / 256, 256, 0, stream>>>(xyz, xyzpad, M * 32);
    cvt(wf,  wf16,  512 * 512);
    cvt(we,  we16,  1024 * 512);
    cvt(wh1, wh116, 256 * 1536);
    cvt(wh2, wh216, 256 * 256);
    cvt(wh3, wh316, 50 * 256);

    // EdgeConv stack (outputs packed into xcat columns: 0,64,128,256)
    edge_layer(xyzpad, 32,  0,   3,   32,  w1, s1, b1, 64,  0);
    edge_layer(xcat,   512, 0,   64,  64,  w2, s2, b2, 64,  64);
    edge_layer(xcat,   512, 64,  64,  64,  w3, s3, b3, 128, 128);
    edge_layer(xcat,   512, 128, 128, 128, w4, s4, b4, 256, 256);

    // x_local -> first 512 cols of xfin (f16)
    gemm16x64_wmma<0, _Float16><<<gemm_grid(512), 32, 0, stream>>>(
        xcat, 512, 0, wf16, 512, xfin, 1536, 0, sf, bf, 512, 512);
    // x_emb (f16)
    gemm16x64_wmma<0, _Float16><<<gemm_grid(1024), 32, 0, stream>>>(
        xfin, 1536, 0, we16, 512, xemb, 1024, 0, se, be, 512, 1024);
    // global max + broadcast into cols 512..1535 of xfin
    colmax_kernel<<<dim3((1024 + 255) / 256, B), 256, 0, stream>>>(xemb, glob, N, 1024);
    const int btot = M * 1024;
    bcast_kernel<<<(btot + 255) / 256, 256, 0, stream>>>(glob, xfin, N, 1536, 512, 1024, btot);
    // head
    gemm16x64_wmma<0, _Float16><<<gemm_grid(256), 32, 0, stream>>>(
        xfin, 1536, 0, wh116, 1536, h1, 256, 0, sh1, bh1, 1536, 256);
    gemm16x64_wmma<0, _Float16><<<gemm_grid(256), 32, 0, stream>>>(
        h1, 256, 0, wh216, 256, h2, 256, 0, sh2, bh2, 256, 256);
    // logits: O=50 padded to one 64-wide tile; W rows clamped, stores guarded
    gemm16x64_wmma<1, float><<<gemm_grid(64), 32, 0, stream>>>(
        h2, 256, 0, wh316, 256, logits, 50, 0, nullptr, bh3, 256, 50);
    (void)in_sizes; (void)n_in; (void)out_size; (void)ws_size;
}
